// BinaryLinear_62818191671354
// MI455X (gfx1250) — matrix operand order repaired and checked by reading, not compiled, not run on hardware
//
#include <hip/hip_runtime.h>
#include <hip/hip_bf16.h>

// CDNA5 / gfx1250 binary-weight linear: out = x @ sign(W)^T + bias
// x: f32 [M=16384, K=4096], W: f32 [N=4096, K=4096], out: f32 [M, N]
// W -> exact bf16 {-1,+1}; x -> bf16 hi/lo split (products with +-1 are exact),
// two v_wmma_f32_16x16x32_bf16 per tile into one f32 accumulator.
// Double-buffered LDS; B tiles staged with global_load_async_to_lds_b128
// (ASYNCcnt) when the toolchain exposes it.

typedef __attribute__((ext_vector_type(16))) __bf16 v16bf;
typedef __attribute__((ext_vector_type(8)))  __bf16 v8bf;
typedef __attribute__((ext_vector_type(8)))  float  v8f;
typedef __attribute__((ext_vector_type(4)))  int    v4i;

#define AS1 __attribute__((address_space(1)))
#define AS3 __attribute__((address_space(3)))

#define K_DIM 4096
#define N_DIM 4096
#define BM 128
#define BN 128
#define BK 32
#define PITCH 40   // 32 data elems + 8 pad -> 80B rows (20 dwords, bank-friendly)

#if defined(__has_builtin)
#if __has_builtin(__builtin_amdgcn_global_load_async_to_lds_b128)
#define HAVE_ASYNC_LDS 1
#endif
#endif
#ifndef HAVE_ASYNC_LDS
#define HAVE_ASYNC_LDS 0
#endif

__device__ __forceinline__ void wait_async_lds() {
#if HAVE_ASYNC_LDS
#if __has_builtin(__builtin_amdgcn_s_wait_asynccnt)
  __builtin_amdgcn_s_wait_asynccnt(0);
#else
  asm volatile("s_wait_asynccnt 0x0" ::: "memory");
#endif
#endif
}

__device__ __forceinline__ unsigned short f32_to_bf16_rne(float f) {
  unsigned u = __float_as_uint(f);
  unsigned r = u + 0x7FFFu + ((u >> 16) & 1u);   // round-to-nearest-even
  return (unsigned short)(r >> 16);
}

__device__ __forceinline__ unsigned short sign_bf16(float f) {
  unsigned u = __float_as_uint(f);
  // sign(w): +1 -> 0x3F80, -1 -> 0xBF80, 0 -> 0x0000 (bf16 bit patterns)
  return (u & 0x7FFFFFFFu) ? (unsigned short)(0x3F80u | ((u >> 16) & 0x8000u))
                           : (unsigned short)0;
}

// ---------------- prepass: binarize W f32[N][K] -> bf16 sign [N][K] ----------
__global__ __launch_bounds__(256) void binarize_weight_kernel(
    const float* __restrict__ w, unsigned short* __restrict__ wb, int total4) {
  int i = blockIdx.x * 256 + threadIdx.x;
  if (i >= total4) return;
  float4 v = ((const float4*)w)[i];
  ushort4 s;
  s.x = sign_bf16(v.x);
  s.y = sign_bf16(v.y);
  s.z = sign_bf16(v.z);
  s.w = sign_bf16(v.w);
  ((ushort4*)wb)[i] = s;
}

// ---------------- main GEMM --------------------------------------------------
// PREBIN=true : B tiles copied from pre-binarized bf16 buffer (async-to-LDS)
// PREBIN=false: B tiles binarized on the fly from f32 W (small-ws fallback)
template <bool PREBIN>
__global__ __launch_bounds__(256) void bin_gemm_kernel(
    const float* __restrict__ x,
    const unsigned short* __restrict__ wb,   // bf16 sign(W), [N][K] (PREBIN)
    const float* __restrict__ wf,            // raw f32 W, [N][K]    (!PREBIN)
    const float* __restrict__ bias,
    float* __restrict__ out) {
  __shared__ unsigned short As_hi[2][BM][PITCH];
  __shared__ unsigned short As_lo[2][BM][PITCH];
  __shared__ unsigned short Bs[2][BN][PITCH];

  const int tid   = threadIdx.x;
  const int lane  = tid & 31;
  const int wave  = tid >> 5;
  const int waveM = wave >> 1;   // 0..3 -> 32-row slabs
  const int waveN = wave & 1;    // 0..1 -> 64-col slabs
  const int l16   = lane & 15;
  const int lhalf = lane >> 4;

  const int gm0 = blockIdx.y * BM;
  const int gn0 = blockIdx.x * BN;   // x fastest-varying -> A strip stays in L2

  // ---- staging helpers ----
  auto stageA = [&](int b, int k0) {
    // 128x32 f32 tile -> bf16 hi/lo; 1024 float4, 8 consecutive threads per row
#pragma unroll
    for (int j = 0; j < 4; ++j) {
      int f4  = tid + j * 256;
      int row = f4 >> 3;
      int col = (f4 & 7) << 2;
      const float* gp = x + (size_t)(gm0 + row) * K_DIM + k0 + col;
      float4 v = *(const float4*)gp;
      if (k0 + 2 * BK < K_DIM) __builtin_prefetch(gp + 2 * BK, 0, 0);
      float vf[4] = {v.x, v.y, v.z, v.w};
      ushort4 h, l;
      unsigned short* hp = &h.x;
      unsigned short* lp = &l.x;
#pragma unroll
      for (int e = 0; e < 4; ++e) {
        unsigned short hh = f32_to_bf16_rne(vf[e]);
        float hf = __uint_as_float(((unsigned)hh) << 16);
        hp[e] = hh;
        lp[e] = f32_to_bf16_rne(vf[e] - hf);
      }
      *(ushort4*)&As_hi[b][row][col] = h;
      *(ushort4*)&As_lo[b][row][col] = l;
    }
  };

  auto stageB = [&](int b, int k0) {
    if constexpr (PREBIN) {
      // 128x32 bf16 tile = 512 x 16B chunks, 4 chunks per row
#pragma unroll
      for (int j = 0; j < 2; ++j) {
        int q   = tid + j * 256;
        int row = q >> 2;
        int col = (q & 3) << 3;
        const unsigned short* gp = wb + (size_t)(gn0 + row) * K_DIM + k0 + col;
        unsigned short* lp = &Bs[b][row][col];
#if HAVE_ASYNC_LDS
        __builtin_amdgcn_global_load_async_to_lds_b128(
            (AS1 v4i*)(uintptr_t)gp,
            (AS3 v4i*)(unsigned)(uintptr_t)lp, 0, 0);
#else
        *(uint4*)lp = *(const uint4*)gp;
#endif
      }
    } else {
      // binarize on the fly from f32 W
#pragma unroll
      for (int j = 0; j < 4; ++j) {
        int f4  = tid + j * 256;
        int row = f4 >> 3;
        int col = (f4 & 7) << 2;
        float4 v = *(const float4*)(wf + (size_t)(gn0 + row) * K_DIM + k0 + col);
        ushort4 s;
        s.x = sign_bf16(v.x); s.y = sign_bf16(v.y);
        s.z = sign_bf16(v.z); s.w = sign_bf16(v.w);
        *(ushort4*)&Bs[b][row][col] = s;
      }
    }
  };

  v8f acc[2][4];
#pragma unroll
  for (int f = 0; f < 2; ++f)
#pragma unroll
    for (int g = 0; g < 4; ++g)
      acc[f][g] = v8f{0.f, 0.f, 0.f, 0.f, 0.f, 0.f, 0.f, 0.f};

  constexpr int NKT = K_DIM / BK;   // 128
  stageB(0, 0);
  stageA(0, 0);

  for (int kt = 0; kt < NKT; ++kt) {
    const int cur = kt & 1;
    if constexpr (PREBIN) wait_async_lds();  // my async B chunks have landed
    __syncthreads();                         // everyone's tile `cur` is ready
    if (kt + 1 < NKT) {                      // stage next tile into other buffer
      stageB(1 - cur, (kt + 1) * BK);
      stageA(1 - cur, (kt + 1) * BK);
    }

    // ---- one K=32 WMMA step from buffer `cur` ----
    v16bf a_hi[2], a_lo[2], bfr[4];
#pragma unroll
    for (int f = 0; f < 2; ++f) {
      // A frag (16x32 bf16, ISA 7.12.2): lane<16 -> row l16, K {c0..c0+7, c0+16..c0+23}
      int row = waveM * 32 + f * 16 + l16;
      int c0  = lhalf * 8;
      v8bf p0 = *(const v8bf*)&As_hi[cur][row][c0];
      v8bf p1 = *(const v8bf*)&As_hi[cur][row][c0 + 16];
      a_hi[f] = __builtin_shufflevector(p0, p1, 0, 1, 2, 3, 4, 5, 6, 7,
                                        8, 9, 10, 11, 12, 13, 14, 15);
      v8bf q0 = *(const v8bf*)&As_lo[cur][row][c0];
      v8bf q1 = *(const v8bf*)&As_lo[cur][row][c0 + 16];
      a_lo[f] = __builtin_shufflevector(q0, q1, 0, 1, 2, 3, 4, 5, 6, 7,
                                        8, 9, 10, 11, 12, 13, 14, 15);
    }
#pragma unroll
    for (int g = 0; g < 4; ++g) {
      // B frag (32x16 bf16): lane holds column n, K {kc..kc+7, kc+16..kc+23}
      int n  = waveN * 64 + g * 16 + l16;
      int kc = lhalf * 8;
      v8bf p0 = *(const v8bf*)&Bs[cur][n][kc];
      v8bf p1 = *(const v8bf*)&Bs[cur][n][kc + 16];
      bfr[g] = __builtin_shufflevector(p0, p1, 0, 1, 2, 3, 4, 5, 6, 7,
                                       8, 9, 10, 11, 12, 13, 14, 15);
    }
#pragma unroll
    for (int f = 0; f < 2; ++f)
#pragma unroll
      for (int g = 0; g < 4; ++g) {
        acc[f][g] = __builtin_amdgcn_wmma_f32_16x16x32_bf16(
            false, a_hi[f], false, bfr[g], (short)0, acc[f][g], false, false);
        acc[f][g] = __builtin_amdgcn_wmma_f32_16x16x32_bf16(
            false, a_lo[f], false, bfr[g], (short)0, acc[f][g], false, false);
      }
  }

  // ---- epilogue: C layout (ISA 7.12.2): VGPR r -> M = r + 8*lhalf, N = l16 ----
#pragma unroll
  for (int g = 0; g < 4; ++g) {
    int nloc = waveN * 64 + g * 16 + l16;
    float bv = bias[gn0 + nloc];
#pragma unroll
    for (int f = 0; f < 2; ++f) {
      int row0 = gm0 + waveM * 32 + f * 16 + lhalf * 8;
#pragma unroll
      for (int r = 0; r < 8; ++r) {
        out[(size_t)(row0 + r) * N_DIM + gn0 + nloc] = acc[f][g][r] + bv;
      }
    }
  }
}

extern "C" void kernel_launch(void* const* d_in, const int* in_sizes, int n_in,
                              void* d_out, int out_size, void* d_ws, size_t ws_size,
                              hipStream_t stream) {
  const float* x    = (const float*)d_in[0];
  const float* w    = (const float*)d_in[1];
  const float* bias = (const float*)d_in[2];
  float* out        = (float*)d_out;

  const int M = in_sizes[0] / K_DIM;        // 16384
  dim3 grid(N_DIM / BN, M / BM);            // n-blocks fastest -> L2 reuse of A

  const size_t wb_bytes = (size_t)N_DIM * K_DIM * sizeof(unsigned short);
  if (ws_size >= wb_bytes) {
    unsigned short* wb = (unsigned short*)d_ws;
    int total4 = (N_DIM * K_DIM) / 4;
    binarize_weight_kernel<<<total4 / 256, 256, 0, stream>>>(w, wb, total4);
    bin_gemm_kernel<true><<<grid, 256, 0, stream>>>(x, wb, nullptr, bias, out);
  } else {
    bin_gemm_kernel<false><<<grid, 256, 0, stream>>>(x, nullptr, w, bias, out);
  }
}
